// MixtureOfDepth_44624710206136
// MI455X (gfx1250) — matrix operand order repaired and checked by reading, not compiled, not run on hardware
//
#include <hip/hip_runtime.h>
#include <hip/hip_bf16.h>
#include <stdint.h>

// ---------------- problem constants (from setup_inputs) ----------------
static constexpr int S = 4096;   // sequence length
static constexpr int B = 4;      // batch
static constexpr int D = 2048;   // d_model
static constexpr int F = 8192;   // d_ff
static constexpr int C = 2048;   // MoD capacity (top-k)

static constexpr int KB1 = D / 32;   // 64  K-tiles of 32 for stage 1
static constexpr int KB2 = F / 32;   // 256 K-tiles of 32 for stage 2
static constexpr int FCH = 128;      // F-chunk width per fused iteration
static constexpr int XSTR = D + 8;   // padded LDS row stride (bf16) for X
static constexpr int GSTR = FCH + 8; // padded LDS row stride (bf16) for G

typedef __bf16 bf16_t;
typedef __attribute__((ext_vector_type(16))) __bf16 v16bf;
typedef __attribute__((ext_vector_type(8)))  __bf16 v8bf;
typedef __attribute__((ext_vector_type(8)))  float  v8f;

static __device__ __forceinline__ v8f wmma_bf16(v16bf a, v16bf b, v8f c) {
    // D = A(16x32 bf16) x B(32x16 bf16) + C(16x16 f32)
    return __builtin_amdgcn_wmma_f32_16x16x32_bf16(
        /*neg_a=*/false, a, /*neg_b=*/false, b,
        /*c_mod=*/(short)0, c, /*reuse_a=*/false, /*reuse_b=*/false);
}

static __device__ __forceinline__ float gelu_tanh(float x) {
    float u = 0.7978845608028654f * (x + 0.044715f * x * x * x);
    return 0.5f * x * (1.0f + tanhf(u));
}

// A-fragment K index pattern (16-bit A, 16x32): e<8 -> K = hi*8+e ; e>=8 -> K = 16+hi*8+(e-8)
// => two contiguous 8-element (16B) LDS reads per lane at k0+hi*8 and k0+16+hi*8.
static __device__ __forceinline__ v16bf load_a_frag(const bf16_t* row, int k0, int hi) {
    v8bf lo = *(const v8bf*)(row + k0 + hi * 8);
    v8bf hi8 = *(const v8bf*)(row + k0 + 16 + hi * 8);
    return __builtin_shufflevector(lo, hi8, 0, 1, 2, 3, 4, 5, 6, 7,
                                   8, 9, 10, 11, 12, 13, 14, 15);
}

// ---------------------------------------------------------------------
// 0) zero per-batch selection counters
__global__ void mod_init_kernel(int* cnt) {
    int t = threadIdx.x;
    if (t < B) cnt[t] = 0;
}

// 1) router logits: logits[b*S+s] = dot(hs[s,b,:], gate_w) + gate_b
//    one wave per token; lanes stride D, shuffle-reduce (wave32).
__global__ void mod_gate_kernel(const float* __restrict__ hs,
                                const float* __restrict__ gw,
                                const float* __restrict__ gb,
                                float* __restrict__ logits) {
    int wid  = threadIdx.x >> 5;
    int lane = threadIdx.x & 31;
    int tok  = blockIdx.x * 8 + wid;          // tok = s*B + b
    if (tok >= S * B) return;
    int s = tok / B, b = tok % B;
    const float* row = hs + ((size_t)s * B + b) * D;
    float sum = 0.f;
    for (int d = lane; d < D; d += 32) sum += row[d] * gw[d];
    #pragma unroll
    for (int off = 16; off > 0; off >>= 1)
        sum += __shfl_xor(sum, off, 32);
    if (lane == 0) logits[b * S + s] = sum + gb[0];
}

// 2) top-k set selection by exact rank (softmax is monotone -> rank logits).
//    tie-break: lower index wins (matches jax.lax.top_k). Order-invariant
//    compaction via atomics is fine: gather/FFN/scatter all use the same idx.
__global__ void mod_topk_kernel(const float* __restrict__ logits,
                                const unsigned char* __restrict__ seq_mask,
                                int* __restrict__ idx_out,
                                int* __restrict__ msk_out,
                                int* __restrict__ cnt) {
    int t = blockIdx.x * blockDim.x + threadIdx.x;
    if (t >= B * S) return;
    int b = t / S, i = t % S;
    const float* lg = logits + (size_t)b * S;
    float v = lg[i];
    int rank = 0;
    for (int j = 0; j < S; ++j) {
        float lj = lg[j];
        rank += (lj > v) || (lj == v && j < i);
    }
    if (rank < C) {
        int pos = atomicAdd(&cnt[b], 1);
        idx_out[b * C + pos] = i;
        msk_out[b * C + pos] = seq_mask[(size_t)b * S + i] ? 1 : 0;
    }
}

// 3) gather selected tokens, convert f32 -> bf16 row-major: Xg[b,c,:]
__global__ void mod_gather_kernel(const float* __restrict__ hs,
                                  const int* __restrict__ idx,
                                  bf16_t* __restrict__ xg) {
    int row = blockIdx.x;            // row in [0, B*C)
    int b = row / C;
    int s = idx[row];
    const float* src = hs + ((size_t)s * B + b) * D;
    bf16_t* dst = xg + (size_t)row * D;
    int d0 = threadIdx.x * 8;        // 256 threads * 8 = 2048
    float4 f0 = *(const float4*)(src + d0);
    float4 f1 = *(const float4*)(src + d0 + 4);
    bf16_t tmp[8] = {(bf16_t)f0.x, (bf16_t)f0.y, (bf16_t)f0.z, (bf16_t)f0.w,
                     (bf16_t)f1.x, (bf16_t)f1.y, (bf16_t)f1.z, (bf16_t)f1.w};
    *(uint4*)(dst + d0) = *(const uint4*)tmp;
}

// 4) convert weights to bf16 and pack into B-fragment tile order:
//    tile (kb,nb) of 32x16 stored as 512 bf16 so the GEMM loads one
//    contiguous 32B v16bf per lane. B layout: N = lane&15, K = e<8 ? (lane>>4)*8 + e : 16 + (lane>>4)*8 + (e-8).
__global__ void mod_packw_kernel(const float* __restrict__ W,
                                 bf16_t* __restrict__ P,
                                 int R, int Cc) {
    __shared__ bf16_t tile_s[32 * 16];
    int KB = R / 32, NB = Cc / 16;
    int ntiles = KB * NB;
    int t = threadIdx.x;             // 512 threads
    for (int tile = blockIdx.x; tile < ntiles; tile += gridDim.x) {
        int kb = tile % KB, nb = tile / KB;
        int r = t >> 4, c = t & 15;
        tile_s[r * 16 + c] = (bf16_t)W[((size_t)(kb * 32 + r)) * Cc + nb * 16 + c];
        __syncthreads();
        int lane = t >> 4, e = t & 15;
        int hi = lane >> 4, l15 = lane & 15;
        P[(size_t)tile * 512 + t] = tile_s[((e & 8) * 2 + hi * 8 + (e & 7)) * 16 + l15];
        __syncthreads();
    }
}

// 5) pass-through: out[:S*B*D] = hs (FFN scatter overwrites selected rows later)
__global__ void mod_copy_kernel(const float* __restrict__ hs,
                                float* __restrict__ out) {
    size_t i = ((size_t)blockIdx.x * blockDim.x + threadIdx.x) * 4;
    *(float4*)(out + i) = *(const float4*)(hs + i);
}

// 5b) second tuple output: sequence_mask appended after the hidden states
__global__ void mod_mask_kernel(const unsigned char* __restrict__ seq_mask,
                                float* __restrict__ out, int tail) {
    int j = blockIdx.x * blockDim.x + threadIdx.x;
    if (j < tail) out[(size_t)S * B * D + j] = seq_mask[j] ? 1.0f : 0.0f;
}

// 6) fused MoD FFN: per 16-token block, Y = gelu(X@W1+b1)@W2+b2 with the
//    F dimension streamed in 128-wide chunks through LDS. 8 waves/WG:
//    stage 1: wave w computes G tile cols [fc+16w, fc+16w+16)  (64 WMMAs)
//    stage 2: wave w accumulates Y cols  [256w, 256w+256)      (64 WMMAs)
__global__ void __launch_bounds__(256)
mod_ffn_kernel(const bf16_t* __restrict__ xg,
               const bf16_t* __restrict__ w1p,
               const bf16_t* __restrict__ w2p,
               const float* __restrict__ b1,
               const float* __restrict__ b2,
               const int* __restrict__ idx,
               const int* __restrict__ msk,
               float* __restrict__ out) {
    extern __shared__ char smem[];
    bf16_t* Xs = (bf16_t*)smem;                         // 16 x XSTR
    bf16_t* Gs = (bf16_t*)(smem + 16 * XSTR * 2);       // 16 x GSTR
    __shared__ int sidx[16];
    __shared__ int smsk[16];

    const int b  = blockIdx.x >> 7;        // C/16 = 128 row-blocks per batch
    const int rb = blockIdx.x & 127;
    const int tid  = threadIdx.x;
    const int wid  = tid >> 5;
    const int lane = tid & 31;
    const int hi   = lane >> 4;
    const int l15  = lane & 15;
    const int m    = l15;                  // A-fragment row for this lane

    if (tid < 16) {
        sidx[tid] = idx[b * C + rb * 16 + tid];
        smsk[tid] = msk[b * C + rb * 16 + tid];
    }
    // load X block (16 x 2048 bf16) into padded LDS
    {
        int d0 = tid * 8;                  // 256 threads x 16B = one row
        for (int mm = 0; mm < 16; ++mm) {
            const bf16_t* src = xg + ((size_t)b * C + rb * 16 + mm) * D;
            *(uint4*)(Xs + mm * XSTR + d0) = *(const uint4*)(src + d0);
        }
    }
    __syncthreads();

    v8f acc[16];
    #pragma unroll
    for (int i = 0; i < 16; ++i) acc[i] = (v8f){};

    const bf16_t* xrow = Xs + m * XSTR;
    const bf16_t* grow = Gs + m * GSTR;

    for (int fc = 0; fc < F; fc += FCH) {
        // ---- stage 1: G = gelu(X @ W1[:, fc+16w .. +16) + b1) ----
        v8f g = (v8f){};
        {
            const bf16_t* w1t = w1p + ((size_t)(fc / 16 + wid) * KB1) * 512 + lane * 16;
            #pragma unroll 4
            for (int kb = 0; kb < KB1; ++kb) {
                v16bf a  = load_a_frag(xrow, kb * 32, hi);
                v16bf bb = *(const v16bf*)(w1t + (size_t)kb * 512);
                __builtin_prefetch(w1t + (size_t)(kb + 8) * 512, 0, 0);
                g = wmma_bf16(a, bb, g);
            }
        }
        {
            float b1v = b1[fc + wid * 16 + l15];
            #pragma unroll
            for (int r = 0; r < 8; ++r) {
                float x = g[r] + b1v;
                Gs[(r + hi * 8) * GSTR + wid * 16 + l15] = (bf16_t)gelu_tanh(x);
            }
        }
        __syncthreads();

        // ---- stage 2: Y[:, 256w .. +256) += G @ W2[fc .. fc+128, :] ----
        #pragma unroll
        for (int nt = 0; nt < 16; ++nt) {
            const bf16_t* w2t =
                w2p + ((size_t)(wid * 16 + nt) * KB2 + (fc / 32)) * 512 + lane * 16;
            #pragma unroll
            for (int kc = 0; kc < 4; ++kc) {
                v16bf a  = load_a_frag(grow, kc * 32, hi);
                v16bf bb = *(const v16bf*)(w2t + (size_t)kc * 512);
                acc[nt] = wmma_bf16(a, bb, acc[nt]);
            }
        }
        __syncthreads();
    }

    // ---- epilogue: +b2, mask, scatter into residual stream (S,B,D) ----
    #pragma unroll
    for (int nt = 0; nt < 16; ++nt) {
        int col = wid * 256 + nt * 16 + l15;
        float b2v = b2[col];
        #pragma unroll
        for (int r = 0; r < 8; ++r) {
            int mo = r + hi * 8;
            float val = smsk[mo] ? (acc[nt][r] + b2v) : 0.0f;
            out[((size_t)sidx[mo] * B + b) * D + col] = val;
        }
    }
}

// ---------------------------------------------------------------------
extern "C" void kernel_launch(void* const* d_in, const int* in_sizes, int n_in,
                              void* d_out, int out_size, void* d_ws, size_t ws_size,
                              hipStream_t stream) {
    const float*         hs   = (const float*)d_in[0];
    const unsigned char* smsk = (const unsigned char*)d_in[1];
    const float*         gw   = (const float*)d_in[2];
    const float*         gb   = (const float*)d_in[3];
    const float*         w1   = (const float*)d_in[4];
    const float*         b1   = (const float*)d_in[5];
    const float*         w2   = (const float*)d_in[6];
    const float*         b2   = (const float*)d_in[7];
    float* out = (float*)d_out;

    // workspace layout (bytes), 256B aligned; total ~96.2 MB
    char* ws = (char*)d_ws;
    int*    ws_idx    = (int*)(ws + 0);                        // B*C*4   = 32 KB
    int*    ws_msk    = (int*)(ws + 32768);                    // B*C*4   = 32 KB
    int*    ws_cnt    = (int*)(ws + 65536);                    // 256 B
    float*  ws_logits = (float*)(ws + 65792);                  // B*S*4   = 64 KB
    bf16_t* ws_xg     = (bf16_t*)(ws + 131328);                // B*C*D*2 = 32 MB
    bf16_t* ws_w1p    = (bf16_t*)(ws + 131328 + 33554432ull);  // D*F*2   = 32 MB
    bf16_t* ws_w2p    = (bf16_t*)(ws + 131328 + 67108864ull);  // F*D*2   = 32 MB

    // router + selection
    mod_init_kernel<<<1, 64, 0, stream>>>(ws_cnt);
    mod_gate_kernel<<<(S * B) / 8, 256, 0, stream>>>(hs, gw, gb, ws_logits);
    mod_topk_kernel<<<(B * S) / 256, 256, 0, stream>>>(ws_logits, smsk,
                                                       ws_idx, ws_msk, ws_cnt);
    // gather + precision/layout conversion (weights land in L2: 64 MB << 192 MB)
    mod_gather_kernel<<<B * C, 256, 0, stream>>>(hs, ws_idx, ws_xg);
    mod_packw_kernel<<<2048, 512, 0, stream>>>(w1, ws_w1p, D, F);
    mod_packw_kernel<<<2048, 512, 0, stream>>>(w2, ws_w2p, F, D);

    // residual pass-through + tuple tail (sequence_mask)
    mod_copy_kernel<<<((size_t)S * B * D) / 4 / 256, 256, 0, stream>>>(hs, out);
    int tail = out_size - S * B * D;
    if (tail > 0)
        mod_mask_kernel<<<(tail + 255) / 256, 256, 0, stream>>>(smsk, out, tail);

    // fused WMMA FFN + scatter
    size_t shmem = (size_t)16 * XSTR * 2 + (size_t)16 * GSTR * 2; // 70144 B
    (void)hipFuncSetAttribute((const void*)mod_ffn_kernel,
                              hipFuncAttributeMaxDynamicSharedMemorySize,
                              (int)shmem);
    mod_ffn_kernel<<<B * (C / 16), 256, shmem, stream>>>(
        ws_xg, ws_w1p, ws_w2p, b1, b2, ws_idx, ws_msk, out);
    (void)n_in; (void)in_sizes; (void)ws_size;
}
